// AttentionScore_78752520340045
// MI455X (gfx1250) — matrix operand order repaired and checked by reading, not compiled, not run on hardware
//
#include <hip/hip_runtime.h>
#include <stdint.h>

typedef __attribute__((ext_vector_type(16))) __bf16 v16bf;
typedef __attribute__((ext_vector_type(8)))  float  v8f;

constexpr int Bn = 4, Hn = 16, Sn = 1024, Rn = 512;
constexpr float SCALE_F = 0.07216878364870322f; // 1/sqrt(64+128)

// Dynamic LDS layout (bytes):
//  sQ   : 16 x 514 bf16  (pad 2 -> 1028B stride, conflict-free)   = 16448
//  sKV  : 64 x 514 bf16                                           = 65792
//  sKVT : 512 x 68 bf16  (pad 4 -> even stride, 4B-aligned b32)   = 69632
//  sS   : 16 x 68 f32 raw scores                                  = 4352
//  sPb  : 16 x 68 bf16 packed probabilities                       = 2176
//  sM/sL/sA : 16 f32 each                                         = 192
constexpr size_t OFF_SQ   = 0;
constexpr size_t OFF_SKV  = 16448;
constexpr size_t OFF_SKVT = 82240;
constexpr size_t OFF_SS   = 151872;
constexpr size_t OFF_SPB  = 156224;
constexpr size_t OFF_SM   = 158400;
constexpr size_t OFF_SL   = 158464;
constexpr size_t OFF_SA   = 158528;
constexpr size_t SMEM_BYTES = 158592;

union Frag {
  v16bf bf;
  uint32_t u[8];
};

__device__ __forceinline__ uint32_t pack2_bf16(float lo, float hi) {
  union { __bf16 h[2]; uint32_t u; } t;
  t.h[0] = (__bf16)lo;
  t.h[1] = (__bf16)hi;
  return t.u;
}
__device__ __forceinline__ uint16_t bf16_bits(float f) {
  union { __bf16 h; uint16_t u; } t;
  t.h = (__bf16)f;
  return t.u;
}

__global__ void __launch_bounds__(128, 2)
mla_attn_kernel(const float* __restrict__ q, const float* __restrict__ kv,
                const float* __restrict__ rope, const int* __restrict__ mask,
                const int* __restrict__ have_causal, float* __restrict__ out) {
  extern __shared__ char smem[];
  uint16_t* sQ   = (uint16_t*)(smem + OFF_SQ);
  uint16_t* sKV  = (uint16_t*)(smem + OFF_SKV);
  uint16_t* sKVT = (uint16_t*)(smem + OFF_SKVT);
  float*    sS   = (float*)(smem + OFF_SS);
  uint16_t* sPb  = (uint16_t*)(smem + OFF_SPB);
  float*    sM   = (float*)(smem + OFF_SM);
  float*    sL   = (float*)(smem + OFF_SL);
  float*    sA   = (float*)(smem + OFF_SA);

  const int tid  = threadIdx.x;
  const int lane = tid & 31;
  const int wave = tid >> 5;   // 4 waves
  const int half = lane >> 4;  // wave32 half (A/B/C layout split)
  const int n16  = lane & 15;

  const int blk = blockIdx.x;  // b*1024 + h*64 + qt
  const int qt = blk & 63;
  const int h  = (blk >> 6) & 15;
  const int b  = blk >> 10;
  const int q0 = qt << 4;

  const bool causal = (have_causal[0] != 0);
  const int kmax = causal ? (q0 + 16) : Sn;

  // ---- stage Q tile (16x512 f32 -> bf16 row-major, stride 514) ----
  const float* qbase = q + (((size_t)(b * Hn + h) * Sn + q0) * Rn);
  {
    const float4* qv = (const float4*)qbase;
    #pragma unroll 2
    for (int i = tid; i < 2048; i += 128) {
      float4 x = qv[i];
      int row = i >> 7, c = (i & 127) << 2;
      uint32_t* d = (uint32_t*)(sQ + row * 514 + c);
      d[0] = pack2_bf16(x.x, x.y);
      d[1] = pack2_bf16(x.z, x.w);
    }
  }
  if (tid < 16) { sM[tid] = -INFINITY; sL[tid] = 0.0f; }
  __syncthreads();

  v8f zero = {};
  v8f O[8];                      // this wave owns out columns [wave*128, wave*128+128)
  #pragma unroll
  for (int t = 0; t < 8; ++t) O[t] = zero;

  for (int k0 = 0; k0 < kmax; k0 += 64) {
    // ---- stage kv tile (64x512) as bf16 into sKV (row-major) and sKVT (transposed)
    const float* kvbase = kv + ((size_t)b * Sn + k0) * Rn;
    {
      const float4* kvv = (const float4*)kvbase;
      #pragma unroll 2
      for (int i = tid; i < 8192; i += 128) {
        float4 x = kvv[i];
        int row = i >> 7, c = (i & 127) << 2;
        uint16_t e0 = bf16_bits(x.x), e1 = bf16_bits(x.y);
        uint16_t e2 = bf16_bits(x.z), e3 = bf16_bits(x.w);
        uint32_t* d = (uint32_t*)(sKV + row * 514 + c);
        d[0] = (uint32_t)e0 | ((uint32_t)e1 << 16);
        d[1] = (uint32_t)e2 | ((uint32_t)e3 << 16);
        sKVT[(c + 0) * 68 + row] = e0;
        sKVT[(c + 1) * 68 + row] = e1;
        sKVT[(c + 2) * 68 + row] = e2;
        sKVT[(c + 3) * 68 + row] = e3;
      }
    }
    if (k0 + 64 < kmax) // pull next kv tile toward L2 (global_prefetch_b8)
      __builtin_prefetch(kvbase + 64 * Rn + tid * 256, 0, 1);
    __syncthreads();

    // ---- GEMM1: nope tile (16q x 16k), K=512 => 16 bf16 WMMAs ----
    // A layout: lane row M=n16; elem j (v=j>>1): Klocal = (v>=4?16:0)+half*8+(v&3)*2+(j&1)
    // B layout: lane col N=n16 (key); elem j (v=j>>1): Klocal = (v>=4?16:0)+half*8+(v&3)*2+(j&1)
    {
      const uint16_t* aRow = sQ + n16 * 514 + half * 8;
      const uint16_t* bRow = sKV + (wave * 16 + n16) * 514 + half * 8;
      v8f Cn = zero;
      #pragma unroll
      for (int f = 0; f < 16; ++f) {
        Frag Af, Bk;
        #pragma unroll
        for (int v = 0; v < 8; ++v)
          Af.u[v] = *(const uint32_t*)(aRow + f * 32 + ((v >= 4) ? 16 : 0) + (v & 3) * 2);
        #pragma unroll
        for (int v = 0; v < 8; ++v)
          Bk.u[v] = *(const uint32_t*)(bRow + f * 32 + ((v >= 4) ? 16 : 0) + (v & 3) * 2);
        Cn = __builtin_amdgcn_wmma_f32_16x16x32_bf16(false, Af.bf, false, Bk.bf,
                                                     (short)0, Cn, false, false);
      }

      // ---- scores = (nope + rope)*scale, mask, spill to sS ----
      const int gkey = k0 + wave * 16 + n16;
      const int km = mask[b * Sn + gkey];
      const float* ropeb = rope + (((size_t)(b * Hn + h) * Sn + q0) * Sn) + gkey;
      #pragma unroll
      for (int v = 0; v < 8; ++v) {
        int M = v + half * 8; // C layout: row = v + 8*half, col = n16
        float s = (Cn[v] + ropeb[(size_t)M * Sn]) * SCALE_F;
        if ((causal && gkey > q0 + M) || km) s = -INFINITY;
        sS[M * 68 + wave * 16 + n16] = s;
      }
    }
    __syncthreads();

    // ---- online softmax stats (one thread per q-row); emit packed bf16 probs ----
    if (tid < 16) {
      const int row = tid;
      const float* sr = sS + row * 68;
      float mold = sM[row];
      float mk = -INFINITY;
      #pragma unroll 8
      for (int k = 0; k < 64; ++k) mk = fmaxf(mk, sr[k]);
      float mnew = fmaxf(mold, mk);
      float alpha = (mnew == -INFINITY) ? 0.0f : __expf(mold - mnew);
      float lsum = 0.0f;
      uint32_t* pr = (uint32_t*)(sPb + row * 68);
      #pragma unroll 8
      for (int k2 = 0; k2 < 32; ++k2) {
        float s0 = sr[2 * k2], s1 = sr[2 * k2 + 1];
        float p0 = (s0 == -INFINITY) ? 0.0f : __expf(s0 - mnew);
        float p1 = (s1 == -INFINITY) ? 0.0f : __expf(s1 - mnew);
        lsum += p0 + p1;
        pr[k2] = pack2_bf16(p0, p1);
      }
      sA[row] = alpha;
      sM[row] = mnew;
      sL[row] = sL[row] * alpha + lsum;
    }
    __syncthreads();

    // ---- rescale accumulators ----
    {
      float al[8];
      #pragma unroll
      for (int v = 0; v < 8; ++v) al[v] = sA[v + half * 8];
      #pragma unroll
      for (int t = 0; t < 8; ++t) {
        #pragma unroll
        for (int v = 0; v < 8; ++v) O[t][v] *= al[v];
      }
    }

    // ---- P A-fragments (16q x 32k bf16), two key-chunks, direct b32 loads ----
    Frag P[2];
    #pragma unroll
    for (int c = 0; c < 2; ++c) {
      #pragma unroll
      for (int v = 0; v < 8; ++v) {
        int kk = c * 32 + ((v >= 4) ? 16 : 0) + half * 8 + (v & 3) * 2;
        P[c].u[v] = *(const uint32_t*)(sPb + n16 * 68 + kk);
      }
    }

    // ---- GEMM2: O += P @ V  (8 r-tiles x 2 key-chunks = 16 WMMAs) ----
    #pragma unroll
    for (int t = 0; t < 8; ++t) {
      const int r0 = wave * 128 + t * 16;
      const uint16_t* vRow = sKVT + (r0 + n16) * 68 + half * 8;
      #pragma unroll
      for (int c = 0; c < 2; ++c) {
        Frag Bv; // B = V[32key x 16r]: lane col N=n16 (r); Klocal = (v>=4?16:0)+half*8+(v&3)*2+(j&1)
        #pragma unroll
        for (int v = 0; v < 8; ++v)
          Bv.u[v] = *(const uint32_t*)(vRow + c * 32 + ((v >= 4) ? 16 : 0) + (v & 3) * 2);
        O[t] = __builtin_amdgcn_wmma_f32_16x16x32_bf16(false, P[c].bf, false, Bv.bf,
                                                       (short)0, O[t], false, false);
      }
    }
    __syncthreads();
  }

  // ---- normalize by row sums and store ----
  float li[8];
  #pragma unroll
  for (int v = 0; v < 8; ++v) li[v] = 1.0f / sL[v + half * 8];
  float* obase = out + (((size_t)(b * Hn + h) * Sn + q0) * Rn);
  #pragma unroll
  for (int t = 0; t < 8; ++t) {
    const int r0 = wave * 128 + t * 16;
    #pragma unroll
    for (int v = 0; v < 8; ++v) {
      obase[(size_t)(v + half * 8) * Rn + r0 + n16] = O[t][v] * li[v];
    }
  }
}

extern "C" void kernel_launch(void* const* d_in, const int* in_sizes, int n_in,
                              void* d_out, int out_size, void* d_ws, size_t ws_size,
                              hipStream_t stream) {
  const float* q    = (const float*)d_in[0]; // (B,H,S,R)
  const float* kvc  = (const float*)d_in[1]; // (B,S,R)
  const float* rope = (const float*)d_in[2]; // (B,H,S,S)
  const int*   mask = (const int*)d_in[3];   // (B,S)
  const int*   hvc  = (const int*)d_in[4];   // scalar
  float* outp = (float*)d_out;               // (B,H,S,R)

  dim3 grid(Bn * Hn * (Sn / 16)); // 4096 blocks: one q-tile each
  dim3 block(128);                // 4 waves
  mla_attn_kernel<<<grid, block, SMEM_BYTES, stream>>>(q, kvc, rope, mask, hvc, outp);
}
